// MSAColumnAttention_63763084476747
// MI455X (gfx1250) — compile-verified, matrix operand order checked by reading, not run on hardware
//
#include <hip/hip_runtime.h>
#include <hip/hip_bf16.h>
#include <math.h>

typedef __attribute__((ext_vector_type(16))) _Float16 v16h;
typedef __attribute__((ext_vector_type(8)))  _Float16 v8h;
typedef __attribute__((ext_vector_type(8)))  float    v8f;

#define N_SEQ 256
#define N_RES 384
#define C_M   256
#define N_HEAD 8
#define C_HEAD 32
#define N_TOK (N_SEQ * N_RES)          // 98304 tokens (r-major: t = r*256 + s)

// ---- workspace layout (in _Float16 elements) ----
// All matrix operands are stored FRAGMENT-PACKED: a 16(rows)x32(k) fragment
// occupies 512 halves; lane L owns 16 contiguous halves at frag + L*16.
#define W_MAT_ELEMS (C_M * C_M)        // 65536
#define WQ_OFF  0
#define WK_OFF  (W_MAT_ELEMS)
#define WV_OFF  (2 * W_MAT_ELEMS)
#define WG_OFF  (3 * W_MAT_ELEMS)
#define WO_OFF  (4 * W_MAT_ELEMS)
#define QKV_ELEMS (N_RES * N_HEAD * N_SEQ * C_HEAD)  // 25165824
#define Q_OFF   (5 * W_MAT_ELEMS)      // packed: [rh][stile(16)][512]
#define K_OFF   (Q_OFF + QKV_ELEMS)    // packed: [rh][stile(16)][512]
#define V_OFF   (K_OFF + QKV_ELEMS)    // packed: [rh][ctile(2)][kk(8)][512]
#define G_OFF   (V_OFF + QKV_ELEMS)    // row-major [t][col] sigmoid gate
#define GO_OFF  (G_OFF + QKV_ELEMS)    // row-major [t][col] raw attn output

// ---------------------------------------------------------------------------
// Fragment helpers (CDNA5 wave32 WMMA 16x16x32 f16 layout, ISA 7.12.2).
//   frag_slot(row16, cc) = slot (lane*16+e) of element (row=row16, k=cc)
//   frag_slot(sl, c) == sl*16 + frag_slot(0, c)   (used to hoist stores)
// ---------------------------------------------------------------------------
__device__ __forceinline__ int frag_slot(int row16, int cc) {
    int hi = (cc >> 3) & 1;
    int e  = (cc & 7) + ((cc >> 4) << 3);
    return ((row16 + (hi << 4)) << 4) + e;
}

// Single contiguous 32B-per-lane read: two adjacent b128 loads, no merges.
__device__ __forceinline__ v16h load_frag_packed(const _Float16* p) {
    return *(const v16h*)(p + (threadIdx.x & 31) * 16);
}

__device__ __forceinline__ v8f wmma_f16(v16h a, v16h b, v8f c) {
    return __builtin_amdgcn_wmma_f32_16x16x32_f16(false, a, false, b, (short)0, c,
                                                  false, false);
}

// ---------------------------------------------------------------------------
// Kernel 0: convert + repack the five f32 weight matrices into f16 fragments.
// ---------------------------------------------------------------------------
__global__ void k0_pack_weights(const float* __restrict__ Wq,
                                const float* __restrict__ Wk,
                                const float* __restrict__ Wv,
                                const float* __restrict__ Wg,
                                const float* __restrict__ Wo,
                                _Float16* __restrict__ ws) {
    int i = blockIdx.x * blockDim.x + threadIdx.x;   // 0 .. 5*65536-1
    if (i >= 5 * W_MAT_ELEMS) return;
    int w = i >> 16;
    int j = i & 65535;
    int k = j >> 8;          // 0..255
    int n = j & 255;         // coalesced across threads
    const float* src = (w == 0) ? Wq : (w == 1) ? Wk : (w == 2) ? Wv
                     : (w == 3) ? Wg : Wo;
    int dst = w * W_MAT_ELEMS + (((n >> 4) * 8 + (k >> 5)) * 512)
            + frag_slot(n & 15, k & 31);
    ws[dst] = (_Float16)src[k * C_M + n];
}

// ---------------------------------------------------------------------------
// Kernel 1: LayerNorm + Q/K/V/G projections.
// grid.x = N_TOK/16 blocks, 256 threads (8 waves), 1 block/WGP reg budget.
// ---------------------------------------------------------------------------
__global__ void __launch_bounds__(256, 1)
k1_ln_proj(const float* __restrict__ m,
           const float* __restrict__ gamma,
           const float* __restrict__ beta,
           const float* __restrict__ bg,
           _Float16* __restrict__ ws) {
    __shared__ __align__(16) float    xs[16][C_M];    // 16 KB
    __shared__ __align__(32) _Float16 xhp[8 * 512];   // 8 KB, fragment-packed

    const int tid = threadIdx.x;
    const int t0  = blockIdx.x * 16;
    const int rB  = t0 >> 8;           // residue index, constant per block
    const int s0  = t0 & 255;          // multiple of 16
    const int stB = s0 >> 4;           // s-tile, constant per block

    // Load 16 tokens as float4 (m is (S, R, C_M))
#pragma unroll
    for (int idx = tid; idx < 16 * 64; idx += 256) {
        int row = idx >> 6, c4 = idx & 63;
        float4 val = ((const float4*)m)[((size_t)(s0 + row) * N_RES + rB) * 64 + c4];
        *(float4*)&xs[row][c4 * 4] = val;
    }
    __syncthreads();

    // LayerNorm: 16 threads per row; write packed fragments
    {
        const int row = tid >> 4, l16 = tid & 15;
        float sum = 0.f;
        for (int c = l16; c < C_M; c += 16) sum += xs[row][c];
#pragma unroll
        for (int mk = 8; mk >= 1; mk >>= 1) sum += __shfl_xor(sum, mk, 32);
        float mean = sum * (1.0f / C_M);
        float vs = 0.f;
        for (int c = l16; c < C_M; c += 16) {
            float d = xs[row][c] - mean; vs += d * d;
        }
#pragma unroll
        for (int mk = 8; mk >= 1; mk >>= 1) vs += __shfl_xor(vs, mk, 32);
        float rstd = rsqrtf(vs * (1.0f / C_M) + 1e-5f);
        for (int c = l16; c < C_M; c += 16) {
            float v = (xs[row][c] - mean) * rstd * gamma[c] + beta[c];
            xhp[(c >> 5) * 512 + frag_slot(row, c & 31)] = (_Float16)v;
        }
    }
    __syncthreads();

    const int wave = tid >> 5;
    const int lane = tid & 31;
    const int nloc = lane & 15;
    const int mhi  = (lane >> 4) << 3;

    // Hoist A fragments (shared by every output tile this wave computes)
    v16h a[8];
#pragma unroll
    for (int kk = 0; kk < 8; ++kk) a[kk] = load_frag_packed(xhp + kk * 512);

#pragma unroll
    for (int mat = 0; mat < 4; ++mat) {          // compile-time constant
        const _Float16* W = (mat == 0) ? (ws + WQ_OFF)
                          : (mat == 1) ? (ws + WK_OFF)
                          : (mat == 2) ? (ws + WV_OFF)
                          :              (ws + WG_OFF);
#pragma unroll
        for (int rep = 0; rep < 2; ++rep) {
            const int nt = wave + rep * 8;
            v16h b[8];
#pragma unroll
            for (int kk = 0; kk < 8; ++kk)
                b[kk] = load_frag_packed(W + (nt * 8 + kk) * 512);
            v8f acc = {};
#pragma unroll
            for (int kk = 0; kk < 8; ++kk) acc = wmma_f16(a[kk], b[kk], acc);

            const int col = nt * 16 + nloc;
            const int h = col >> 5, c = col & 31;
            const int rh = rB * N_HEAD + h;
            if (mat == 0) {
                _Float16* qp = ws + Q_OFF + ((size_t)rh * 16 + stB) * 512
                             + frag_slot(0, c) + mhi * 16;
#pragma unroll
                for (int v = 0; v < 8; ++v)
                    qp[v * 16] = (_Float16)(acc[v] * 0.17677669529663689f);
            } else if (mat == 1) {
                _Float16* kp = ws + K_OFF + ((size_t)rh * 16 + stB) * 512
                             + frag_slot(0, c) + mhi * 16;
#pragma unroll
                for (int v = 0; v < 8; ++v)
                    kp[v * 16] = (_Float16)acc[v];
            } else if (mat == 2) {
                // V frag: rows = c (2 tiles of 16), k-dim = s (8 tiles of 32)
                const int kkv   = s0 >> 5;
                const int sbase = (s0 & 31) + mhi;   // 0/8/16/24
                _Float16* vp = ws + V_OFF
                             + (((size_t)rh * 2 + (c >> 4)) * 8 + kkv) * 512;
#pragma unroll
                for (int v = 0; v < 8; ++v)
                    vp[frag_slot(c & 15, sbase + v)] = (_Float16)acc[v];
            } else {
                const float bgv = bg[col];
                _Float16* gp = ws + G_OFF + (size_t)(t0 + mhi) * C_M + col;
#pragma unroll
                for (int v = 0; v < 8; ++v) {
                    float x = acc[v] + bgv;
                    gp[v * C_M] = (_Float16)(1.0f / (1.0f + __expf(-x)));
                }
            }
        }
    }
}

// ---------------------------------------------------------------------------
// Kernel 2: attention per (r, h, q-half). 256 threads = 8 waves, each wave
// owns one 16-row m-tile of the 128 query rows. Dynamic LDS:
//   float lg[128*256] (128 KB) + packed attn atp[64*512 halves] (64 KB).
// ---------------------------------------------------------------------------
__global__ void __launch_bounds__(256, 1)
k2_attention(_Float16* __restrict__ ws) {
    extern __shared__ __align__(32) char smem[];
    float*    lg  = (float*)smem;
    _Float16* atp = (_Float16*)(smem + 128 * 256 * sizeof(float));

    const int bx = blockIdx.x;
    const int rh = bx >> 1;            // 0..3071
    const int qhalf = bx & 1;
    const int r = rh >> 3, h = rh & 7;

    const _Float16* Qw = ws + Q_OFF + (size_t)rh * 16 * 512;
    const _Float16* Kw = ws + K_OFF + (size_t)rh * 16 * 512;
    const _Float16* Vw = ws + V_OFF + (size_t)rh * 16 * 512;
    _Float16*       GO = ws + GO_OFF;

    const int tid  = threadIdx.x;
    const int wave = tid >> 5;
    const int lane = tid & 31;
    const int nloc = lane & 15;
    const int mhi  = (lane >> 4) << 3;

    // ---- logits = (Q/sqrt(c)) @ K^T : all operands fragment-packed ----
    {
        v16h qa = load_frag_packed(Qw + (qhalf * 8 + wave) * 512);
#pragma unroll
        for (int g = 0; g < 4; ++g) {
            v16h kb[4];
#pragma unroll
            for (int i = 0; i < 4; ++i)
                kb[i] = load_frag_packed(Kw + (g * 4 + i) * 512);
#pragma unroll
            for (int i = 0; i < 4; ++i) {
                v8f acc = {};
                acc = wmma_f16(qa, kb[i], acc);
                const int nt = g * 4 + i;
                float* lp = lg + (wave * 16 + mhi) * 256 + nt * 16 + nloc;
#pragma unroll
                for (int v = 0; v < 8; ++v)
                    lp[v * 256] = acc[v];
            }
        }
    }
    __syncthreads();

    // ---- softmax: one thread per query row; write packed attn frags ----
    if (tid < 128) {
        float* row = lg + tid * 256;
        const float4* row4 = (const float4*)row;
        float mx = -1e30f;
#pragma unroll 8
        for (int c = 0; c < 64; ++c) {
            float4 x = row4[c];
            mx = fmaxf(mx, fmaxf(fmaxf(x.x, x.y), fmaxf(x.z, x.w)));
        }
        float sum = 0.f;
#pragma unroll 8
        for (int c = 0; c < 256; ++c) {
            float e = __expf(row[c] - mx); row[c] = e; sum += e;
        }
        float inv = 1.0f / sum;
        const int tile = tid >> 4, rloc = tid & 15;
#pragma unroll
        for (int c8 = 0; c8 < 32; ++c8) {
            int cc = (c8 * 8) & 31;
            int kk = c8 >> 2;
            int hi = (cc >> 3) & 1;
            int eb = (cc >> 4) << 3;             // 0 or 8
            v8h chunk;
#pragma unroll
            for (int e = 0; e < 8; ++e)
                chunk[e] = (_Float16)(row[c8 * 8 + e] * inv);
            *(v8h*)(atp + ((tile * 8 + kk) * 512)
                        + ((rloc + (hi << 4)) << 4) + eb) = chunk;
        }
    }
    __syncthreads();

    // ---- o = attn @ V : packed frags ----
    {
        v8f acc0 = {}, acc1 = {};
#pragma unroll
        for (int kp = 0; kp < 4; ++kp) {         // kk pairs
            v16h b00 = load_frag_packed(Vw + (0 * 8 + kp * 2 + 0) * 512);
            v16h b01 = load_frag_packed(Vw + (0 * 8 + kp * 2 + 1) * 512);
            v16h b10 = load_frag_packed(Vw + (1 * 8 + kp * 2 + 0) * 512);
            v16h b11 = load_frag_packed(Vw + (1 * 8 + kp * 2 + 1) * 512);
            v16h a0 = load_frag_packed(atp + (wave * 8 + kp * 2 + 0) * 512);
            v16h a1 = load_frag_packed(atp + (wave * 8 + kp * 2 + 1) * 512);
            acc0 = wmma_f16(a0, b00, acc0);
            acc1 = wmma_f16(a0, b10, acc1);
            acc0 = wmma_f16(a1, b01, acc0);
            acc1 = wmma_f16(a1, b11, acc1);
        }
        _Float16* gp = GO + ((size_t)r * N_SEQ + qhalf * 128 + wave * 16 + mhi) * C_M
                     + h * 32 + nloc;
#pragma unroll
        for (int v = 0; v < 8; ++v) {
            gp[v * C_M]      = (_Float16)acc0[v];
            gp[v * C_M + 16] = (_Float16)acc1[v];
        }
    }
}

// ---------------------------------------------------------------------------
// Kernel 3: out = (g*o) @ Wo + bo -> f32 in (S, R, C_M) layout.
// Gate fused during packed-LDS staging (coalesced v8h loads).
// grid.x = N_TOK/16, 256 threads.
// ---------------------------------------------------------------------------
__global__ void __launch_bounds__(256, 1)
k3_out_proj(const _Float16* __restrict__ ws,
            const float* __restrict__ bo,
            float* __restrict__ out) {
    __shared__ __align__(32) _Float16 gop[8 * 512];   // 8 KB, fragment-packed

    const int tid = threadIdx.x;
    const int t0  = blockIdx.x * 16;
    const int rB  = t0 >> 8;
    const int s0  = t0 & 255;
    const _Float16* GO = ws + GO_OFF;
    const _Float16* Gw = ws + G_OFF;
    const _Float16* Wo = ws + WO_OFF;

    // Stage g*o into packed LDS (512 v8h chunks; global side coalesced)
#pragma unroll
    for (int idx = tid; idx < 16 * C_M / 8; idx += 256) {
        v8h o = *(const v8h*)(GO + (size_t)t0 * C_M + idx * 8);
        v8h g = *(const v8h*)(Gw + (size_t)t0 * C_M + idx * 8);
        v8h rr;
#pragma unroll
        for (int e = 0; e < 8; ++e)
            rr[e] = (_Float16)((float)o[e] * (float)g[e]);
        int row = idx >> 5, c8 = idx & 31;
        int cc = (c8 * 8) & 31;
        int kk = c8 >> 2;
        int hi = (cc >> 3) & 1;
        int eb = (cc >> 4) << 3;
        *(v8h*)(gop + kk * 512 + ((row + (hi << 4)) << 4) + eb) = rr;
    }
    __syncthreads();

    const int wave = tid >> 5;
    const int lane = tid & 31;
    const int nloc = lane & 15;
    const int mhi  = (lane >> 4) << 3;

    // Hoist A fragments once
    v16h a[8];
#pragma unroll
    for (int kk = 0; kk < 8; ++kk) a[kk] = load_frag_packed(gop + kk * 512);

#pragma unroll
    for (int half = 0; half < 2; ++half) {
        int nt = wave + half * 8;
        v16h b[8];
#pragma unroll
        for (int kk = 0; kk < 8; ++kk)
            b[kk] = load_frag_packed(Wo + (nt * 8 + kk) * 512);
        v8f acc = {};
#pragma unroll
        for (int kk = 0; kk < 8; ++kk) acc = wmma_f16(a[kk], b[kk], acc);

        int col = nt * 16 + nloc;
        float bias = bo[col];
        float* op = out + ((size_t)(s0 + mhi) * N_RES + rB) * C_M + col;
#pragma unroll
        for (int v = 0; v < 8; ++v)
            op[(size_t)v * N_RES * C_M] = acc[v] + bias;
    }
}

// ---------------------------------------------------------------------------
extern "C" void kernel_launch(void* const* d_in, const int* in_sizes, int n_in,
                              void* d_out, int out_size, void* d_ws, size_t ws_size,
                              hipStream_t stream) {
    const float* m     = (const float*)d_in[0];
    const float* gamma = (const float*)d_in[1];
    const float* beta  = (const float*)d_in[2];
    const float* Wq    = (const float*)d_in[3];
    const float* Wk    = (const float*)d_in[4];
    const float* Wv    = (const float*)d_in[5];
    const float* Wg    = (const float*)d_in[6];
    const float* bg    = (const float*)d_in[7];
    const float* Wo    = (const float*)d_in[8];
    const float* bo    = (const float*)d_in[9];
    float* out = (float*)d_out;
    _Float16* ws = (_Float16*)d_ws;

    // 0) weights -> f16 fragments
    {
        int n = 5 * W_MAT_ELEMS;
        k0_pack_weights<<<(n + 255) / 256, 256, 0, stream>>>(Wq, Wk, Wv, Wg, Wo, ws);
    }
    // 1) LayerNorm + QKVG projections
    k1_ln_proj<<<N_TOK / 16, 256, 0, stream>>>(m, gamma, beta, bg, ws);
    // 2) attention (r,h) x 2 query halves; 192 KB dynamic LDS
    k2_attention<<<N_RES * N_HEAD * 2, 256, 128 * 256 * 6, stream>>>(ws);
    // 3) output projection
    k3_out_proj<<<N_TOK / 16, 256, 0, stream>>>(ws, bo, out);
}
